// _RecurrentBase_76441827934538
// MI455X (gfx1250) — compile-verified, matrix operand order checked by reading, not run on hardware
//
#include <hip/hip_runtime.h>
#include <hip/hip_bf16.h>

// ---------------------------------------------------------------------------
// CDNA5 (gfx1250) bf16 WMMA implementation of the recurrent encoder/LSTM/decoder
// ---------------------------------------------------------------------------
typedef __attribute__((ext_vector_type(16))) __bf16 v16bf;
typedef __attribute__((ext_vector_type(8)))  float  v8f;

union FragU {
    v16bf v;
    uint4 q[2];
};

// D = A(16x32 bf16) * B(32x16 bf16) + C (16x16 f32)
__device__ __forceinline__ v8f wmma_bf16(v16bf a, v16bf b, v8f c) {
    return __builtin_amdgcn_wmma_f32_16x16x32_bf16(
        /*neg_a=*/false, a, /*neg_b=*/false, b,
        /*c_mod=*/(short)0, c, /*reuse_a=*/false, /*reuse_b=*/false);
}

// A-fragment load from an LDS row-major 16 x K slab (KLD = padded stride).
// Lane mapping (16-bit A 16x32): lanes 0-15 row M=lane hold K = k0+[0..7],[16..23];
// lanes 16-31 row M=lane-16 hold K = k0+[8..15],[24..31].
__device__ __forceinline__ v16bf load_a_frag(const __bf16* ldsA, int KLD, int kb, int lane) {
    int row = lane & 15;
    int seg = (lane >> 4) * 8;
    const __bf16* p = ldsA + row * KLD + kb * 32 + seg;
    FragU u;
    u.q[0] = *(const uint4*)(p);
    u.q[1] = *(const uint4*)(p + 16);
    return u.v;
}

// B-fragment load from packed (fragment-major) weights: one 32B load per lane.
__device__ __forceinline__ v16bf load_b_frag(const __bf16* wpk, int nTiles, int kb, int nb, int lane) {
    return *(const v16bf*)(wpk + (((size_t)kb * nTiles + nb) * 32 + lane) * 16);
}

// ---------------------------------------------------------------------------
// Weight packing: fp32 KxN row-major -> bf16 WMMA-fragment-major
// packed[((kb*nTiles + nb)*32 + lane)*16 + idx] = W[k, n]
//   klocal = (idx&7) + ((idx>>3)<<4) + ((lane>>4)<<3),  n = nb*16 + (lane&15)
// ---------------------------------------------------------------------------
__global__ void pack_w_kernel(const float* __restrict__ W, __bf16* __restrict__ Wpk,
                              int K, int NOUT) {
    int total = K * NOUT;
    int nTiles = NOUT >> 4;
    for (int i = blockIdx.x * blockDim.x + threadIdx.x; i < total;
         i += gridDim.x * blockDim.x) {
        int idx  = i & 15;
        int rest = i >> 4;
        int lane = rest & 31; rest >>= 5;
        int nb = rest % nTiles;
        int kb = rest / nTiles;
        int klocal = (idx & 7) + ((idx >> 3) << 4) + ((lane >> 4) << 3);
        int n = nb * 16 + (lane & 15);
        int k = kb * 32 + klocal;
        Wpk[i] = (__bf16)W[(size_t)k * NOUT + n];
    }
}

__global__ void cvt_f32_bf16_kernel(const float* __restrict__ src,
                                    __bf16* __restrict__ dst, int n) {
    for (int i = blockIdx.x * blockDim.x + threadIdx.x; i < n;
         i += gridDim.x * blockDim.x)
        dst[i] = (__bf16)src[i];
}

// ---------------------------------------------------------------------------
// Fused GEMM + bias + LayerNorm (+ optional LeakyReLU), bf16 out.
// Block = 16 rows x NOUT cols, 256 threads (8 waves), TPW = NOUT/128 tiles/wave.
// ---------------------------------------------------------------------------
template <int NOUT, bool LEAKY>
__global__ __launch_bounds__(256) void gemm_ln_kernel(
    const __bf16* __restrict__ A, const __bf16* __restrict__ Wpk,
    const float* __restrict__ bias, const float* __restrict__ gamma,
    const float* __restrict__ beta, __bf16* __restrict__ out, int K) {
    constexpr int TPW = NOUT / 128;
    constexpr int NT  = NOUT / 16;
    __shared__ __bf16 ldsA[16 * 520];
    __shared__ float  ldsC[16 * NOUT];
    __shared__ float  redS[256], redQ[256], rowMu[16], rowRs[16];

    const int KLD = K + 8;
    const size_t m0 = (size_t)blockIdx.x * 16;

    // cooperative A-slab load: 16 rows x K bf16
    for (int idx = threadIdx.x; idx < 16 * (K >> 3); idx += 256) {
        int r  = idx / (K >> 3);
        int c8 = (idx % (K >> 3)) * 8;
        *(uint4*)&ldsA[r * KLD + c8] = *(const uint4*)&A[(m0 + r) * K + c8];
    }
    __syncthreads();

    int wave = threadIdx.x >> 5, lane = threadIdx.x & 31;
    v8f acc[TPW];
    for (int j = 0; j < TPW; ++j)
        for (int v = 0; v < 8; ++v) acc[j][v] = 0.f;

    for (int kb = 0; kb < (K >> 5); ++kb) {
        v16bf a = load_a_frag(ldsA, KLD, kb, lane);
        for (int j = 0; j < TPW; ++j) {
            v16bf b = load_b_frag(Wpk, NT, kb, wave * TPW + j, lane);
            acc[j] = wmma_bf16(a, b, acc[j]);
        }
    }

    // accumulators -> LDS fp32 slab (+bias). C layout: lanes<16 M=v, lanes>=16 M=v+8
    int nLane = lane & 15, mBase = (lane >> 4) * 8;
    for (int j = 0; j < TPW; ++j) {
        int c0 = (wave * TPW + j) * 16;
        for (int v = 0; v < 8; ++v)
            ldsC[(mBase + v) * NOUT + c0 + nLane] = acc[j][v] + bias[c0 + nLane];
    }
    __syncthreads();

    // LayerNorm: 16 threads per row
    int r = threadIdx.x >> 4, s = threadIdx.x & 15;
    float sum = 0.f, sq = 0.f;
    for (int c = s; c < NOUT; c += 16) {
        float v = ldsC[r * NOUT + c];
        sum += v; sq += v * v;
    }
    redS[threadIdx.x] = sum; redQ[threadIdx.x] = sq;
    __syncthreads();
    if (s == 0) {
        float S = 0.f, Q = 0.f;
        for (int i = 0; i < 16; ++i) { S += redS[r * 16 + i]; Q += redQ[r * 16 + i]; }
        float mu  = S / NOUT;
        float var = Q / NOUT - mu * mu;
        rowMu[r] = mu;
        rowRs[r] = rsqrtf(var + 1e-5f);
    }
    __syncthreads();
    for (int idx = threadIdx.x; idx < 16 * NOUT; idx += 256) {
        int rr = idx / NOUT, cc = idx % NOUT;
        float v = (ldsC[idx] - rowMu[rr]) * rowRs[rr] * gamma[cc] + beta[cc];
        if (LEAKY) v = (v >= 0.f) ? v : 0.01f * v;
        out[(m0 + rr) * NOUT + cc] = (__bf16)v;
    }
}

// ---------------------------------------------------------------------------
// Plain GEMM + bias, 32-row blocks (two M-tiles share each B fragment, halving
// the dominant L2 weight traffic). Tile-looped for low register pressure.
// ---------------------------------------------------------------------------
template <int NOUT, bool OUTF32>
__global__ __launch_bounds__(256) void gemm_store32_kernel(
    const __bf16* __restrict__ A, const __bf16* __restrict__ Wpk,
    const float* __restrict__ bias, void* __restrict__ outv, int K) {
    constexpr int NT = NOUT / 16;
    __shared__ __bf16 ldsA[32 * 520];
    const int KLD = K + 8;
    const size_t m0 = (size_t)blockIdx.x * 32;

    for (int idx = threadIdx.x; idx < 32 * (K >> 3); idx += 256) {
        int r  = idx / (K >> 3);
        int c8 = (idx % (K >> 3)) * 8;
        *(uint4*)&ldsA[r * KLD + c8] = *(const uint4*)&A[(m0 + r) * K + c8];
    }
    __syncthreads();

    int wave = threadIdx.x >> 5, lane = threadIdx.x & 31;
    int nLane = lane & 15, mBase = (lane >> 4) * 8;

    for (int nb = wave; nb < NT; nb += 8) {
        v8f acc0, acc1;
        for (int v = 0; v < 8; ++v) { acc0[v] = 0.f; acc1[v] = 0.f; }
        for (int kb = 0; kb < (K >> 5); ++kb) {
            v16bf b  = load_b_frag(Wpk, NT, kb, nb, lane);
            v16bf a0 = load_a_frag(ldsA, KLD, kb, lane);
            v16bf a1 = load_a_frag(ldsA + 16 * KLD, KLD, kb, lane);
            acc0 = wmma_bf16(a0, b, acc0);
            acc1 = wmma_bf16(a1, b, acc1);
        }
        int c = nb * 16 + nLane;
        float bv = bias[c];
        for (int v = 0; v < 8; ++v) {
            float v0 = acc0[v] + bv;
            float v1 = acc1[v] + bv;
            size_t o0 = (m0 + mBase + v) * NOUT + c;
            size_t o1 = (m0 + 16 + mBase + v) * NOUT + c;
            if (OUTF32) { ((float*)outv)[o0] = v0; ((float*)outv)[o1] = v1; }
            else { ((__bf16*)outv)[o0] = (__bf16)v0; ((__bf16*)outv)[o1] = (__bf16)v1; }
        }
    }
}

// ---------------------------------------------------------------------------
// One LSTM timestep: g = gates_x[:,t,:] + (m*h_prev) @ w_hh, then cell update.
// Grid = 64 blocks: (row-block rb 0..15) x (column slice cs 0..3). Each wave
// owns ONE 16-col tile per gate group at matching offsets, so i/f/g/o for a
// cell share the same lane/VGPR slot -> pure-register cell update, and the
// serial per-wave WMMA chain is only 64 instructions per step.
// h is double-buffered (h_prev read-only, h_next write-only) because column
// slices of the same rows live in different workgroups.
// ---------------------------------------------------------------------------
__global__ __launch_bounds__(256) void lstm_step_kernel(
    const __bf16* __restrict__ gates,   // (N*T, 2048), row = n*512 + t
    const int*    __restrict__ mask,    // (N, T)
    const __bf16* __restrict__ whh_pk,  // packed 512x2048
    const float* __restrict__ h_prev, float* __restrict__ h_next,
    float* __restrict__ c_state,        // (256, 512)
    __bf16* __restrict__ hs,            // (N*T, 512)
    int t) {
    __shared__ __bf16 ldsA[16 * 520];
    const int KLD = 520;
    const int rb = blockIdx.x >> 2;   // row block 0..15
    const int cs = blockIdx.x & 3;    // column slice 0..3
    const int n0 = rb * 16;

    // masked h_prev -> bf16 LDS slab (16 x 512)
    for (int idx = threadIdx.x; idx < 16 * 128; idx += 256) {
        int r  = idx >> 7;
        int c4 = (idx & 127) * 4;
        float m = (float)mask[(n0 + r) * 512 + t];
        float4 hv = *(const float4*)&h_prev[(size_t)(n0 + r) * 512 + c4];
        __bf16* dst = &ldsA[r * KLD + c4];
        dst[0] = (__bf16)(hv.x * m);
        dst[1] = (__bf16)(hv.y * m);
        dst[2] = (__bf16)(hv.z * m);
        dst[3] = (__bf16)(hv.w * m);
    }
    __syncthreads();

    int wave = threadIdx.x >> 5, lane = threadIdx.x & 31;
    v8f acc[4];
    for (int g = 0; g < 4; ++g)
        for (int v = 0; v < 8; ++v) acc[g][v] = 0.f;

    const int NT = 128;  // 2048/16 column tiles
    for (int kb = 0; kb < 16; ++kb) {
        v16bf a = load_a_frag(ldsA, KLD, kb, lane);
        for (int g = 0; g < 4; ++g) {
            int nb = g * 32 + cs * 8 + wave;
            v16bf b = load_b_frag(whh_pk, NT, kb, nb, lane);
            acc[g] = wmma_bf16(a, b, acc[g]);
        }
    }

    // fused cell update, entirely from accumulator registers
    int nLane = lane & 15, mBase = (lane >> 4) * 8;
    int col = cs * 128 + wave * 16 + nLane;
    for (int v = 0; v < 8; ++v) {
        int nB = n0 + mBase + v;
        size_t grow = ((size_t)nB * 512 + t) * 2048;
        float gi = (float)gates[grow + col]        + acc[0][v];
        float gf = (float)gates[grow + 512 + col]  + acc[1][v];
        float gg = (float)gates[grow + 1024 + col] + acc[2][v];
        float go = (float)gates[grow + 1536 + col] + acc[3][v];
        float i_ = 1.f / (1.f + __expf(-gi));
        float f_ = 1.f / (1.f + __expf(-gf));
        float g_ = tanhf(gg);
        float o_ = 1.f / (1.f + __expf(-go));
        float m  = (float)mask[nB * 512 + t];
        size_t so = (size_t)nB * 512 + col;
        float c_old = c_state[so] * m;
        float c_new = f_ * c_old + i_ * g_;
        float h_new = o_ * tanhf(c_new);
        c_state[so] = c_new;
        h_next[so]  = h_new;
        hs[((size_t)nB * 512 + t) * 512 + col] = (__bf16)h_new;
    }
}

// ---------------------------------------------------------------------------
extern "C" void kernel_launch(void* const* d_in, const int* in_sizes, int n_in,
                              void* d_out, int out_size, void* d_ws, size_t ws_size,
                              hipStream_t stream) {
    (void)in_sizes; (void)n_in; (void)out_size; (void)ws_size;
    const float* x      = (const float*)d_in[0];
    const int*   mask   = (const int*)d_in[1];
    const float* enc_w1 = (const float*)d_in[2];
    const float* enc_b1 = (const float*)d_in[3];
    const float* enc_g1 = (const float*)d_in[4];
    const float* enc_be1= (const float*)d_in[5];
    const float* enc_w2 = (const float*)d_in[6];
    const float* enc_b2 = (const float*)d_in[7];
    const float* enc_g2 = (const float*)d_in[8];
    const float* enc_be2= (const float*)d_in[9];
    const float* w_ih   = (const float*)d_in[10];
    const float* w_hh   = (const float*)d_in[11];
    const float* b_lstm = (const float*)d_in[12];
    const float* dec_w1 = (const float*)d_in[13];
    const float* dec_b1 = (const float*)d_in[14];
    const float* dec_g1 = (const float*)d_in[15];
    const float* dec_be1= (const float*)d_in[16];
    const float* dec_w2 = (const float*)d_in[17];
    const float* dec_b2 = (const float*)d_in[18];

    const size_t M = 131072;  // N*T tokens
    char* ws = (char*)d_ws;
    size_t off = 0;
    auto take = [&](size_t b) { size_t o = off; off += (b + 255) & ~(size_t)255; return o; };
    __bf16* x_bf   = (__bf16*)(ws + take(M * 256 * 2));   // later reused as dec1 out
    __bf16* bufA   = (__bf16*)(ws + take(M * 512 * 2));   // h_enc, later hs
    __bf16* bufB   = (__bf16*)(ws + take(M * 512 * 2));   // feat
    __bf16* gates  = (__bf16*)(ws + take(M * 2048 * 2));
    __bf16* wpk_e1 = (__bf16*)(ws + take(256 * 512 * 2));
    __bf16* wpk_e2 = (__bf16*)(ws + take(512 * 512 * 2));
    __bf16* wpk_ih = (__bf16*)(ws + take(512 * 2048 * 2));
    __bf16* wpk_hh = (__bf16*)(ws + take(512 * 2048 * 2));
    __bf16* wpk_d1 = (__bf16*)(ws + take(512 * 256 * 2));
    __bf16* wpk_d2 = (__bf16*)(ws + take(256 * 64 * 2));
    float*  h_st0  = (float*)(ws + take(256 * 512 * 4));
    float*  h_st1  = (float*)(ws + take(256 * 512 * 4));
    float*  c_st   = (float*)(ws + take(256 * 512 * 4));
    __bf16* hs   = bufA;   // h_enc dead before LSTM writes
    __bf16* dbuf = x_bf;   // x_bf dead after enc1

    hipMemsetAsync(h_st0, 0, 256 * 512 * 4, stream);
    hipMemsetAsync(h_st1, 0, 256 * 512 * 4, stream);
    hipMemsetAsync(c_st,  0, 256 * 512 * 4, stream);

    cvt_f32_bf16_kernel<<<2048, 256, 0, stream>>>(x, x_bf, (int)(M * 256));
    pack_w_kernel<<<512, 256, 0, stream>>>(enc_w1, wpk_e1, 256, 512);
    pack_w_kernel<<<512, 256, 0, stream>>>(enc_w2, wpk_e2, 512, 512);
    pack_w_kernel<<<1024, 256, 0, stream>>>(w_ih, wpk_ih, 512, 2048);
    pack_w_kernel<<<1024, 256, 0, stream>>>(w_hh, wpk_hh, 512, 2048);
    pack_w_kernel<<<512, 256, 0, stream>>>(dec_w1, wpk_d1, 512, 256);
    pack_w_kernel<<<128, 256, 0, stream>>>(dec_w2, wpk_d2, 256, 64);

    const int MB16 = (int)(M / 16);  // 8192 row-blocks
    const int MB32 = (int)(M / 32);  // 4096 row-blocks
    // encoder: Linear+LN+LeakyReLU, Linear+LN
    gemm_ln_kernel<512, true><<<MB16, 256, 0, stream>>>(
        x_bf, wpk_e1, enc_b1, enc_g1, enc_be1, bufA, 256);
    gemm_ln_kernel<512, false><<<MB16, 256, 0, stream>>>(
        bufA, wpk_e2, enc_b2, enc_g2, enc_be2, bufB, 512);
    // gates_x = feat @ w_ih + b_lstm
    gemm_store32_kernel<2048, false><<<MB32, 256, 0, stream>>>(
        bufB, wpk_ih, b_lstm, (void*)gates, 512);
    // sequential LSTM scan with ping-pong h state
    for (int t = 0; t < 512; ++t) {
        const float* hp = (t & 1) ? h_st1 : h_st0;
        float*       hn = (t & 1) ? h_st0 : h_st1;
        lstm_step_kernel<<<64, 256, 0, stream>>>(gates, mask, wpk_hh, hp, hn, c_st, hs, t);
    }
    // decoder: Linear+LN+LeakyReLU, Linear
    gemm_ln_kernel<256, true><<<MB16, 256, 0, stream>>>(
        hs, wpk_d1, dec_b1, dec_g1, dec_be1, dbuf, 512);
    gemm_store32_kernel<64, true><<<MB32, 256, 0, stream>>>(
        dbuf, wpk_d2, dec_b2, d_out, 256);
}
